// NCA_7559142440962
// MI455X (gfx1250) — matrix operand order repaired and checked by reading, not compiled, not run on hardware
//
#include <hip/hip_runtime.h>
#include <hip/hip_bf16.h>

typedef __attribute__((ext_vector_type(16))) _Float16 v16h;
typedef __attribute__((ext_vector_type(2)))  __fp16   v2hf;   // return type of cvt_pkrtz
typedef __attribute__((ext_vector_type(8)))  float    v8f;

#define SEQ_L        16384
#define NCH          32
#define POS_PER_BLK  512
#define THREADS      256
#define NWAVES       8
#define STRIDE_H     48              // halves per position row in activation tiles (16B aligned, bank-spread)

// ---------------------------------------------------------------------------
// Weight prep: pack w1/w2/w3 (f32) into f16 WMMA A-fragments in d_ws.
// contiguous-k A-fragment order used here for V_WMMA_F32_16X16X32_F16 (16-bit A 16x32, wave32):
//   lane 0-15:  M=lane,    VGPR0..7 hold K=0..15
//   lane 16-31: M=lane-16, VGPR0..7 hold K=16..31
// half index j in v16h: v=j>>1, s=j&1 ->  K = (g<<4) + (v<<1) + s
// Fragment ids: 0..5 = layer1 (mt*3+t), 6..7 = layer2 (mt), 8..9 = layer3 (mt)
// ---------------------------------------------------------------------------
__global__ void nca_prep_weights(const float* __restrict__ w1,
                                 const float* __restrict__ w2,
                                 const float* __restrict__ w3,
                                 _Float16* __restrict__ wf) {
  for (int i = threadIdx.x; i < 10 * 512; i += blockDim.x) {
    int frag = i >> 9;
    int lane = (i >> 4) & 31;
    int j    = i & 15;
    int m    = lane & 15;
    int g    = lane >> 4;
    int v    = j >> 1, s = j & 1;
    int k    = (g << 4) + (v << 1) + s;   // 0..31 within K-tile
    float val;
    if (frag < 6) {                    // layer 1: K-tile t = tap, k = input channel
      int mt = frag / 3, t = frag - mt * 3;
      int o  = mt * 16 + m;
      val = w1[(o * NCH + k) * 3 + t];
    } else if (frag < 8) {             // layer 2 (1x1)
      int o = (frag - 6) * 16 + m;
      val = w2[o * NCH + k];
    } else {                           // layer 3 (1x1)
      int o = (frag - 8) * 16 + m;
      val = w3[o * NCH + k];
    }
    wf[i] = (_Float16)val;
  }
}

// stage one accumulator row group (8 channels) into an activation tile, f32->f16
__device__ __forceinline__ void stage_act(const v8f& acc, const float* bias,
                                          _Float16* dst) {
#pragma unroll
  for (int r = 0; r < 8; ++r) {
    float v = acc[r] + bias[r];
    v = v > 0.0f ? v : 0.0f;           // ReLU
    dst[r] = (_Float16)v;
  }
}

__device__ __forceinline__ v8f wmma_f16(const v16h a, const v16h b, const v8f c) {
  return __builtin_amdgcn_wmma_f32_16x16x32_f16(false, a, false, b, (short)0, c,
                                                false, false);
}

// ---------------------------------------------------------------------------
// Fused 3-layer kernel. Block = 8 waves x 32 lanes, covers 512 positions of one
// batch row. x staged once to LDS as f16 [col][ch]; activations bounce through
// per-wave LDS tiles to re-shape C/D fragments into next-layer B fragments.
// ---------------------------------------------------------------------------
__global__ __launch_bounds__(THREADS)
void nca_fused(const float* __restrict__ x,
               const float* __restrict__ b1,
               const float* __restrict__ b2,
               const float* __restrict__ b3,
               const _Float16* __restrict__ wf,
               float* __restrict__ out) {
  __shared__ __align__(32) _Float16 xt[(POS_PER_BLK + 2) * NCH];      // [col][ch]
  __shared__ __align__(32) _Float16 ht1[NWAVES * 16 * STRIDE_H];      // [wave][pos][ch]
  __shared__ __align__(32) _Float16 ht2[NWAVES * 16 * STRIDE_H];

  const int tid   = threadIdx.x;
  const int blk   = blockIdx.x;
  const int batch = blk >> 5;                 // 64 batches
  const int chunk = blk & 31;                 // 32 chunks of 512 positions
  const int p0    = chunk * POS_PER_BLK;

  const size_t xbase = (size_t)batch * NCH * SEQ_L;

  // ---- stage x tile: columns p0-1 .. p0+512 (circular), f32 -> packed f16 ----
  // process channel PAIRS so each iteration is 2 coalesced b32 loads ->
  // one v_cvt_pk_rtz_f16_f32 -> one ds_store_b32 (channel pair contiguous in [col][ch])
  for (int i = tid; i < (POS_PER_BLK + 2) * (NCH / 2); i += THREADS) {
    int c2 = i / (POS_PER_BLK + 2);           // channel pair 0..15
    int j  = i - c2 * (POS_PER_BLK + 2);      // column 0..513
    int gp = (p0 + j - 1 + SEQ_L) & (SEQ_L - 1);
    float lo = x[xbase + (size_t)(2 * c2) * SEQ_L + gp];
    float hi = x[xbase + (size_t)(2 * c2 + 1) * SEQ_L + gp];
    v2hf pk = __builtin_amdgcn_cvt_pkrtz(lo, hi);
    *(v2hf*)(xt + j * NCH + 2 * c2) = pk;
  }

  // speculative prefetch of next chunk's x rows into L2 (global_prefetch_b8)
  {
    int p0n = ((chunk + 1) & 31) * POS_PER_BLK;
    const float* np = x + xbase + (size_t)(tid & 31) * SEQ_L + p0n + (tid >> 5) * 64;
    __builtin_prefetch(np, 0, 1);
  }

  const int lane = tid & 31;
  const int wave = tid >> 5;
  const int g    = lane >> 4;                 // lane half-group
  const int n    = lane & 15;                 // N (position) within tile

  // ---- load A fragments (pre-packed) ----
  v16h A1[2][3], A2[2], A3[2];
#pragma unroll
  for (int mt = 0; mt < 2; ++mt) {
#pragma unroll
    for (int t = 0; t < 3; ++t)
      A1[mt][t] = *(const v16h*)(wf + (mt * 3 + t) * 512 + lane * 16);
    A2[mt] = *(const v16h*)(wf + (6 + mt) * 512 + lane * 16);
    A3[mt] = *(const v16h*)(wf + (8 + mt) * 512 + lane * 16);
  }

  // ---- per-lane biases: C/D fragment row r -> channel mt*16 + g*8 + r ----
  float bb1[2][8], bb2[2][8], bb3[2][8];
#pragma unroll
  for (int mt = 0; mt < 2; ++mt)
#pragma unroll
    for (int r = 0; r < 8; ++r) {
      int o = mt * 16 + g * 8 + r;
      bb1[mt][r] = b1[o];
      bb2[mt][r] = b2[o];
      bb3[mt][r] = b3[o];
    }

  __syncthreads();

  _Float16* myh1 = ht1 + wave * 16 * STRIDE_H;
  _Float16* myh2 = ht2 + wave * 16 * STRIDE_H;

  for (int nt = 0; nt < 4; ++nt) {
    const int base = wave * 64 + nt * 16;     // position offset in block
    // tap t reads x column (base + n + t) == position (p0+base+n) + t - 1

    // ---- layer 1: y = sum_t W_t * X_shift(t), K = 3x32 ----
    v8f acc0 = {}, acc1 = {};
#pragma unroll
    for (int t = 0; t < 3; ++t) {
      const v16h bf = *(const v16h*)(xt + (base + n + t) * NCH + g * 16);
      acc0 = wmma_f16(A1[0][t], bf, acc0);
      acc1 = wmma_f16(A1[1][t], bf, acc1);
    }
    stage_act(acc0, bb1[0], myh1 + n * STRIDE_H + g * 8);
    stage_act(acc1, bb1[1], myh1 + n * STRIDE_H + 16 + g * 8);

    // ---- layer 2: 1x1, K = 32 ----
    {
      const v16h bf = *(const v16h*)(myh1 + n * STRIDE_H + g * 16);
      v8f z = {};
      v8f a0 = wmma_f16(A2[0], bf, z);
      v8f a1 = wmma_f16(A2[1], bf, z);
      stage_act(a0, bb2[0], myh2 + n * STRIDE_H + g * 8);
      stage_act(a1, bb2[1], myh2 + n * STRIDE_H + 16 + g * 8);
    }

    // ---- layer 3: 1x1 + fast sigmoid (v_exp + v_rcp), direct f32 store ----
    {
      const v16h bf = *(const v16h*)(myh2 + n * STRIDE_H + g * 16);
      v8f z = {};
      v8f a0 = wmma_f16(A3[0], bf, z);
      v8f a1 = wmma_f16(A3[1], bf, z);
      const int p = p0 + base + n;
#pragma unroll
      for (int r = 0; r < 8; ++r) {
        float v0 = a0[r] + bb3[0][r];
        float v1 = a1[r] + bb3[1][r];
        v0 = __builtin_amdgcn_rcpf(1.0f + __expf(-v0));   // sigmoid, 2 trans ops
        v1 = __builtin_amdgcn_rcpf(1.0f + __expf(-v1));
        out[xbase + (size_t)(g * 8 + r) * SEQ_L + p] = v0;
        out[xbase + (size_t)(16 + g * 8 + r) * SEQ_L + p] = v1;
      }
    }
  }
}

extern "C" void kernel_launch(void* const* d_in, const int* in_sizes, int n_in,
                              void* d_out, int out_size, void* d_ws, size_t ws_size,
                              hipStream_t stream) {
  const float* x  = (const float*)d_in[0];
  const float* w1 = (const float*)d_in[1];
  const float* b1 = (const float*)d_in[2];
  const float* w2 = (const float*)d_in[3];
  const float* b2 = (const float*)d_in[4];
  const float* w3 = (const float*)d_in[5];
  const float* b3 = (const float*)d_in[6];
  _Float16* wf    = (_Float16*)d_ws;     // 10 fragments * 512 halves = 10 KB

  nca_prep_weights<<<1, THREADS, 0, stream>>>(w1, w2, w3, wf);

  const int nblocks = 64 * (SEQ_L / POS_PER_BLK);   // 2048
  nca_fused<<<nblocks, THREADS, 0, stream>>>(x, b1, b2, b3, wf, (float*)d_out);
}
